// OptimizedLSTM_71554155152219
// MI455X (gfx1250) — matrix operand order repaired and checked by reading, not compiled, not run on hardware
//
#include <hip/hip_runtime.h>

// ---------------------------------------------------------------------------
// 2-layer LSTM for MI455X (gfx1250, wave32, WMMA bf16 16x16x32).
//  - Weights stacked [W_in; W_hi] (2048 x 4096), converted to bf16 and
//    pre-packed into the exact WMMA B-fragment lane layout (32B/lane/tile).
//  - Per timestep: gates = [x_t | h_{t-1}] @ [W_in; W_hi] + bias, fused with
//    the sigmoid/tanh cell update. f32 accumulation via
//    v_wmma_f32_16x16x32_bf16. The 16MB weight block stays L2-resident.
//  - Inner K-loop: 3-buffer, copy-free software pipeline (rotation period ==
//    buffer count). Each WMMA group runs with ~2 slices of loads in flight.
//  - 512 sequential launches per layer express the recurrence (graph-safe).
// ---------------------------------------------------------------------------

typedef __bf16 bf16_t;
typedef __attribute__((ext_vector_type(16))) __bf16 v16bf;
typedef __attribute__((ext_vector_type(8)))  __bf16 v8bf;
typedef __attribute__((ext_vector_type(8)))  float  v8f;

#define SEQ  512
#define BS   64
#define HID  1024
#define GATE 4096        // 4*HID
#define KTOT 2048        // HID (input) + HID (recurrent)

__device__ __forceinline__ float sigmoid_f(float x) {
    return 1.0f / (1.0f + __expf(-x));
}
// tanh via the hardware exp transcendental: tanh(x) = 1 - 2/(e^{2x}+1)
__device__ __forceinline__ float tanh_f(float x) {
    return 1.0f - 2.0f / (__expf(2.0f * x) + 1.0f);
}

// --- zero initial h (bf16) and cell states (f32) ---------------------------
__global__ void lstm_init(bf16_t* h0, float* c1, float* c2) {
    int i = blockIdx.x * blockDim.x + threadIdx.x;
    if (i < BS * HID) {
        h0[i] = (bf16_t)0.0f;
        c1[i] = 0.0f;
        c2[i] = 0.0f;
    }
}

// --- f32 -> bf16 bulk convert (for x) --------------------------------------
__global__ void cvt_f32_bf16(const float* __restrict__ src,
                             bf16_t* __restrict__ dst, int n) {
    int i = blockIdx.x * blockDim.x + threadIdx.x;
    if (i < n) dst[i] = (bf16_t)src[i];
}

// --- pack stacked [W_in; W_hi] into WMMA B-fragment layout -----------------
// Bpack linear index = ((kt*256 + ntile_g)*32 + lane)*16 + e
//   lanes 0-15 : B[k0 + (e<8 ? e : 8+e)   ][n0 + lane]      (e = 0..15)
//   lanes 16-31: B[k0 + (e<8 ? 8+e : 16+e)][n0 + lane - 16]
__global__ void pack_weights(const float* __restrict__ Win,
                             const float* __restrict__ Whi,
                             bf16_t* __restrict__ Bpack) {
    int idx = blockIdx.x * blockDim.x + threadIdx.x;   // 0 .. 2048*4096-1
    if (idx >= KTOT * GATE) return;
    int e    = idx & 15;
    int lane = (idx >> 4) & 31;
    int ntg  = (idx >> 9) & 255;
    int kt   = idx >> 17;
    int K = kt * 32 + ((lane < 16) ? 0 : 8) + ((e < 8) ? e : 8 + e);
    int N = ntg * 16 + (lane & 15);
    float w = (K < HID) ? Win[K * GATE + N] : Whi[(K - HID) * GATE + N];
    Bpack[idx] = (bf16_t)w;
}

// --- fragment bundle for one K-slice ---------------------------------------
struct Frag {
    v16bf a, b0, b1, b2, b3;
};

__device__ __forceinline__ void load_frags(const bf16_t* __restrict__ ap,
                                           const bf16_t* __restrict__ bp,
                                           Frag& f) {
    union { v16bf v; v8bf h[2]; } A;
    A.h[0] = *(const v8bf*)(ap);
    A.h[1] = *(const v8bf*)(ap + 16);
    f.a  = A.v;
    f.b0 = *(const v16bf*)(bp);
    f.b1 = *(const v16bf*)(bp +  64 * 512);
    f.b2 = *(const v16bf*)(bp + 128 * 512);
    f.b3 = *(const v16bf*)(bp + 192 * 512);
}

// A-row slice selector: slices 0..31 from x_t, 32..63 from h_{t-1}.
__device__ __forceinline__ const bf16_t* aslice(const bf16_t* __restrict__ ax,
                                                const bf16_t* __restrict__ ah,
                                                int k) {
    return (k < 32) ? (ax + k * 32) : (ah + (k - 32) * 32);
}

#define WMMA4(F)                                                               \
    do {                                                                       \
        acc0 = __builtin_amdgcn_wmma_f32_16x16x32_bf16(false, (F).a, false,    \
                   (F).b0, (short)0, acc0, false, false);                      \
        acc1 = __builtin_amdgcn_wmma_f32_16x16x32_bf16(false, (F).a, false,    \
                   (F).b1, (short)0, acc1, false, false);                      \
        acc2 = __builtin_amdgcn_wmma_f32_16x16x32_bf16(false, (F).a, false,    \
                   (F).b2, (short)0, acc2, false, false);                      \
        acc3 = __builtin_amdgcn_wmma_f32_16x16x32_bf16(false, (F).a, false,    \
                   (F).b3, (short)0, acc3, false, false);                      \
    } while (0)

// --- one fused LSTM timestep ----------------------------------------------
// 256 wave tasks: task = m_tile (4) x n_tile (64). Each wave computes the
// 16x16 tile of all four gates (i,f,g,o) at hidden cols [n0, n0+16), then
// does the cell update for those elements.
__global__ void __launch_bounds__(256)
lstm_step(const bf16_t* __restrict__ Ax,     // [BS, HID] bf16 layer input at t
          const bf16_t* __restrict__ Hprev,  // [BS, HID] bf16 h_{t-1}
          const bf16_t* __restrict__ Bpack,  // packed [64][256][32][16]
          const float*  __restrict__ bias,   // [GATE]
          float*        __restrict__ C,      // [BS, HID] cell state (in/out)
          bf16_t*       __restrict__ Hout,   // [BS, HID] bf16 h_t
          float*        __restrict__ HoutF,  // [BS, HID] f32 h_t (or null)
          int writeF) {
    const int lane  = threadIdx.x & 31;
    const int wave  = threadIdx.x >> 5;
    const int task  = blockIdx.x * 8 + wave;   // 0..255
    const int m0    = (task & 3) * 16;
    const int ntile = task >> 2;               // 0..63
    const int row   = m0 + (lane & 15);        // batch row for A fragment
    const int koff  = (lane < 16) ? 0 : 8;     // K-half split per A layout

    // Row base pointers for the two K-phases (x part, then h part).
    const bf16_t* arow_x = Ax    + row * HID + koff;
    const bf16_t* arow_h = Hprev + row * HID + koff;
    const bf16_t* bbase  = Bpack + (size_t)ntile * 512 + (size_t)lane * 16;

    v8f acc0 = {}, acc1 = {}, acc2 = {}, acc3 = {};

    // Copy-free 3-buffer pipeline: prologue fills slices 0..2.
    Frag f0, f1, f2;
    load_frags(aslice(arow_x, arow_h, 0), bbase,                           f0);
    load_frags(aslice(arow_x, arow_h, 1), bbase + (size_t)1 * (256 * 512), f1);
    load_frags(aslice(arow_x, arow_h, 2), bbase + (size_t)2 * (256 * 512), f2);

    // Steady state: consume kt..kt+2, reload the just-consumed buffer 3 ahead.
    for (int kt = 0; kt < 60; kt += 3) {
        __builtin_prefetch(bbase + (size_t)(kt + 6) * (256 * 512), 0, 3);
        WMMA4(f0);
        load_frags(aslice(arow_x, arow_h, kt + 3),
                   bbase + (size_t)(kt + 3) * (256 * 512), f0);
        WMMA4(f1);
        load_frags(aslice(arow_x, arow_h, kt + 4),
                   bbase + (size_t)(kt + 4) * (256 * 512), f1);
        WMMA4(f2);
        load_frags(aslice(arow_x, arow_h, kt + 5),
                   bbase + (size_t)(kt + 5) * (256 * 512), f2);
    }
    // Epilogue: f0=60, f1=61, f2=62; slice 63 reloads f0 after its consume.
    WMMA4(f0);
    load_frags(aslice(arow_x, arow_h, 63),
               bbase + (size_t)63 * (256 * 512), f0);
    WMMA4(f1);
    WMMA4(f2);
    WMMA4(f0);

    // C/D layout: element v of lane L -> row = m0 + v + (L<16 ? 0 : 8),
    //                                    col = n0 + (L & 15)
    const int col   = ntile * 16 + (lane & 15);
    const int rbase = m0 + ((lane < 16) ? 0 : 8);
    const float bi = bias[col];
    const float bf = bias[HID + col];
    const float bg = bias[2 * HID + col];
    const float bo = bias[3 * HID + col];

    #pragma unroll
    for (int v = 0; v < 8; ++v) {
        const int r = rbase + v;
        float gi = sigmoid_f(acc0[v] + bi);
        float gf = sigmoid_f(acc1[v] + bf);
        float gg = tanh_f(acc2[v] + bg);
        float go = sigmoid_f(acc3[v] + bo);
        float c  = gf * C[r * HID + col] + gi * gg;
        C[r * HID + col] = c;
        float h = go * tanh_f(c);
        Hout[r * HID + col] = (bf16_t)h;
        if (writeF) HoutF[r * HID + col] = h;
    }
}

// --- copy final (h, c) into the tail of d_out ------------------------------
__global__ void lstm_tail(float* __restrict__ out, const float* __restrict__ c2) {
    int i = blockIdx.x * blockDim.x + threadIdx.x;
    if (i < BS * HID) {
        const size_t hsN = (size_t)SEQ * BS * HID;
        out[hsN + i]            = out[(size_t)(SEQ - 1) * BS * HID + i];
        out[hsN + BS * HID + i] = c2[i];
    }
}

extern "C" void kernel_launch(void* const* d_in, const int* in_sizes, int n_in,
                              void* d_out, int out_size, void* d_ws, size_t ws_size,
                              hipStream_t stream) {
    const float* x    = (const float*)d_in[0];
    const float* Wii0 = (const float*)d_in[1];
    const float* Wii  = (const float*)d_in[2];
    const float* Whi  = (const float*)d_in[3];
    const float* bias = (const float*)d_in[4];
    float* out = (float*)d_out;

    // Workspace carve-up (~225 MB).
    char* p = (char*)d_ws;
    bf16_t* xbf = (bf16_t*)p;  p += (size_t)SEQ * BS * HID * 2;   // 64 MB
    bf16_t* hs1 = (bf16_t*)p;  p += (size_t)SEQ * BS * HID * 2;   // 64 MB
    bf16_t* hs2 = (bf16_t*)p;  p += (size_t)SEQ * BS * HID * 2;   // 64 MB
    bf16_t* Bp0 = (bf16_t*)p;  p += (size_t)KTOT * GATE * 2;      // 16 MB
    bf16_t* Bp1 = (bf16_t*)p;  p += (size_t)KTOT * GATE * 2;      // 16 MB
    bf16_t* h0  = (bf16_t*)p;  p += (size_t)BS * HID * 2;
    float*  c1  = (float*)p;   p += (size_t)BS * HID * 4;
    float*  c2  = (float*)p;   p += (size_t)BS * HID * 4;

    lstm_init<<<dim3((BS * HID + 255) / 256), dim3(256), 0, stream>>>(h0, c1, c2);

    {
        int n = SEQ * BS * HID;
        cvt_f32_bf16<<<dim3((n + 255) / 256), dim3(256), 0, stream>>>(x, xbf, n);
    }
    {
        int n = KTOT * GATE;
        pack_weights<<<dim3(n / 256), dim3(256), 0, stream>>>(Wii0, Whi, Bp0);
        pack_weights<<<dim3(n / 256), dim3(256), 0, stream>>>(Wii,  Whi, Bp1);
    }

    // Layer 1: x -> hs1 (bf16)
    for (int t = 0; t < SEQ; ++t) {
        const bf16_t* Ax = xbf + (size_t)t * BS * HID;
        const bf16_t* Hp = (t == 0) ? h0 : hs1 + (size_t)(t - 1) * BS * HID;
        bf16_t* Ho = hs1 + (size_t)t * BS * HID;
        lstm_step<<<dim3(32), dim3(256), 0, stream>>>(
            Ax, Hp, Bp0, bias, c1, Ho, (float*)nullptr, 0);
    }
    // Layer 2: hs1 -> hs2 (bf16) + f32 output into d_out
    for (int t = 0; t < SEQ; ++t) {
        const bf16_t* Ax = hs1 + (size_t)t * BS * HID;
        const bf16_t* Hp = (t == 0) ? h0 : hs2 + (size_t)(t - 1) * BS * HID;
        bf16_t* Ho = hs2 + (size_t)t * BS * HID;
        float*  Hf = out + (size_t)t * BS * HID;
        lstm_step<<<dim3(32), dim3(256), 0, stream>>>(
            Ax, Hp, Bp1, bias, c2, Ho, Hf, 1);
    }

    lstm_tail<<<dim3((BS * HID + 255) / 256), dim3(256), 0, stream>>>(out, c2);
}
